// MatrixLoss_12000138625280
// MI455X (gfx1250) — matrix operand order repaired and checked by reading, not compiled, not run on hardware
//
#include <hip/hip_runtime.h>

typedef __attribute__((ext_vector_type(16))) __bf16 v16bf;
typedef __attribute__((ext_vector_type(8)))  float  v8f;

#define ROWSTRIDE 144          // 64 bf16 = 128B + 16B pad (16B aligned, conflict-free)
#define NROWS     384
#define NEGF      (-1e30f)

// LDS union region: staging (384*144 = 55296 B) then six score matrices (61440 B)
#define SMEM_BYTES 61440
#define BIG_OFF(q)   ((q) * 16384)          // 3 x 64x64 f32
#define SMALL_OFF(q) (49152 + (q) * 4096)   // 3 x 32x32 f32

__device__ __forceinline__ unsigned short f2bf(float f) {
    unsigned int u = __float_as_uint(f);
    unsigned int r = u + 0x7FFFu + ((u >> 16) & 1u);   // round-to-nearest-even
    return (unsigned short)(r >> 16);
}

// A fragment: 16x32 bf16 (ISA 7.12.2): lane L -> row M=L%16, K halves split by L/16
__device__ __forceinline__ v16bf frag_A(const unsigned char* base, int lane, int k0) {
    int r = lane & 15, g = lane >> 4;
    const unsigned char* row = base + r * ROWSTRIDE;
    union { uint4 u[2]; v16bf v; } u;
    u.u[0] = *(const uint4*)(row + (k0 + 8 * g) * 2);        // K = k0+8g .. +7
    u.u[1] = *(const uint4*)(row + (k0 + 16 + 8 * g) * 2);   // K = k0+16+8g .. +7
    return u.v;
}

// B fragment: 32x16 bf16: lane L -> column N=L%16; K = k0+8*(L/16) .. +7, then k0+16+8*(L/16) .. +7
__device__ __forceinline__ v16bf frag_B(const unsigned char* base, int lane, int k0) {
    int r = lane & 15, g = lane >> 4;
    const unsigned char* row = base + r * ROWSTRIDE;
    union { uint4 u[2]; v16bf v; } u;
    u.u[0] = *(const uint4*)(row + (k0 + 8 * g) * 2);
    u.u[1] = *(const uint4*)(row + (k0 + 16 + 8 * g) * 2);
    return u.v;
}

// tile index (0..59) -> LDS strip row bases for A and B operands
__device__ __forceinline__ void tile_rows(int t, int& aRow, int& bRow) {
    if (t < 48) {
        int q = t >> 4, u = t & 15, i = u >> 2, j = u & 3;
        aRow = (q == 2 ? 64 : 0) + 16 * i;          // obj_p @0, obj_n @64
        bRow = (q == 1 ? 192 : 128) + 16 * j;       // sent_p @128, sent_n @192
    } else {
        int u = t - 48, q = u >> 2, v2 = u & 3, i = v2 >> 1, j = v2 & 1;
        aRow = (q == 2 ? 288 : 256) + 16 * i;       // pred_p @256, pred_n @288
        bRow = (q == 1 ? 352 : 320) + 16 * j;       // rels_p @320, rels_n @352
    }
}

__global__ __launch_bounds__(256)
void matrix_loss_kernel(const float* __restrict__ obj_p,  const float* __restrict__ pred_p,
                        const int*   __restrict__ n_obj_p, const int* __restrict__ n_pred_p,
                        const float* __restrict__ sent_p, const float* __restrict__ rels_p,
                        const int*   __restrict__ len_sp,  const int* __restrict__ n_rels_p,
                        const float* __restrict__ obj_n,  const float* __restrict__ pred_n,
                        const int*   __restrict__ n_obj_n, const int* __restrict__ n_pred_n,
                        const float* __restrict__ sent_n, const float* __restrict__ rels_n,
                        const int*   __restrict__ len_sn,  const int* __restrict__ n_rels_n,
                        float* __restrict__ out)
{
    __shared__ __align__(16) unsigned char smem[SMEM_BYTES];
    __shared__ float s_scratch[256];
    __shared__ float s_res[12];
    __shared__ int   s_offs[6];

    const int b    = blockIdx.x;
    const int tid  = threadIdx.x;
    const int lane = tid & 31;
    const int w    = tid >> 5;

    const int cnt_obj_p  = n_obj_p[b];
    const int cnt_pred_p = n_pred_p[b];
    const int cnt_rels_p = n_rels_p[b];
    const int cnt_obj_n  = n_obj_n[b];
    const int cnt_pred_n = n_pred_n[b];
    const int cnt_rels_n = n_rels_n[b];
    const int lsp = len_sp[b];
    const int lsn = len_sn[b];

    // ---- ragged offsets: prefix sums of 6 count arrays up to b ----
    {
        const int* cntArrs[6] = { n_obj_p, n_pred_p, n_rels_p, n_obj_n, n_pred_n, n_rels_n };
        int* scr = (int*)s_scratch;
        for (int a = 0; a < 6; ++a) {
            scr[tid] = (tid < b) ? cntArrs[a][tid] : 0;
            __syncthreads();
            for (int s = 128; s > 0; s >>= 1) {
                if (tid < s) scr[tid] += scr[tid + s];
                __syncthreads();
            }
            if (tid == 0) s_offs[a] = scr[0];
            __syncthreads();
        }
    }
    const int off_obj_p  = s_offs[0], off_pred_p = s_offs[1], off_rels_p = s_offs[2];
    const int off_obj_n  = s_offs[3], off_pred_n = s_offs[4], off_rels_n = s_offs[5];

    // ---- per-wave tile assignment, padded to 64 tiles so the K-loop is branch-free ----
    // Real tiles t in [0,60); tiles 60..63 (waves 4..7, slot 7) alias tiles 0..3 and
    // are computed redundantly but never written back. ~6.7% extra WMMA, zero branches.
    int aOff[8], bOff[8];
    #pragma unroll
    for (int ti = 0; ti < 8; ++ti) {
        int t = w + ti * 8;
        int tf = (t < 60) ? t : (t - 60);
        int aRow, bRow;
        tile_rows(tf, aRow, bRow);
        aOff[ti] = aRow * ROWSTRIDE;
        bOff[ti] = bRow * ROWSTRIDE;
    }

    v8f acc[8] = {};

    // ---- main K loop: stream D=1024 in 16 chunks of 64 ----
    const int seg = tid & 15;
    const int col = seg << 2;             // 4 floats per thread per row
    for (int kc = 0; kc < 16; ++kc) {
        const int kbase = kc * 64;
        // stage 384 rows x 64 cols as bf16 into LDS (coalesced float4 reads)
        for (int rr = tid >> 4; rr < NROWS; rr += 16) {
            const float* src = nullptr;
            bool valid = true;
            if (rr < 64)        { int rl = rr;       valid = rl < cnt_obj_p;  src = obj_p  + (size_t)(off_obj_p  + rl) * 1024; }
            else if (rr < 128)  { int rl = rr - 64;  valid = rl < cnt_obj_n;  src = obj_n  + (size_t)(off_obj_n  + rl) * 1024; }
            else if (rr < 192)  { int rl = rr - 128; src = sent_p + ((size_t)b * 64 + rl) * 1024; }
            else if (rr < 256)  { int rl = rr - 192; src = sent_n + ((size_t)b * 64 + rl) * 1024; }
            else if (rr < 288)  { int rl = rr - 256; valid = rl < cnt_pred_p; src = pred_p + (size_t)(off_pred_p + rl) * 1024; }
            else if (rr < 320)  { int rl = rr - 288; valid = rl < cnt_pred_n; src = pred_n + (size_t)(off_pred_n + rl) * 1024; }
            else if (rr < 352)  { int rl = rr - 320; valid = rl < cnt_rels_p; src = rels_p + (size_t)(off_rels_p + rl) * 1024; }
            else                { int rl = rr - 352; valid = rl < cnt_rels_n; src = rels_n + (size_t)(off_rels_n + rl) * 1024; }
            float4 v = make_float4(0.f, 0.f, 0.f, 0.f);
            if (valid) v = *(const float4*)(src + kbase + col);
            uint2 packed;
            packed.x = (unsigned int)f2bf(v.x) | ((unsigned int)f2bf(v.y) << 16);
            packed.y = (unsigned int)f2bf(v.z) | ((unsigned int)f2bf(v.w) << 16);
            *(uint2*)(smem + rr * ROWSTRIDE + col * 2) = packed;
        }
        __syncthreads();

        // straight-line compute: 8 tiles x 2 WMMAs, no EXEC manipulation
        #pragma unroll
        for (int ti = 0; ti < 8; ++ti) {
            #pragma unroll
            for (int k0 = 0; k0 < 64; k0 += 32) {
                v16bf va = frag_A(smem + aOff[ti], lane, k0);
                v16bf vb = frag_B(smem + bOff[ti], lane, k0);
                acc[ti] = __builtin_amdgcn_wmma_f32_16x16x32_bf16(
                    false, va, false, vb, (short)0, acc[ti], false, false);
            }
        }
        __syncthreads();
    }

    // ---- write real tiles to LDS score matrices (union with staging region) ----
    #pragma unroll
    for (int ti = 0; ti < 8; ++ti) {
        int t = w + ti * 8;
        if (t < 60) {
            float* Sb; int ld, i, j;
            if (t < 48) { int q = t >> 4, u = t & 15; i = u >> 2; j = u & 3;
                          Sb = (float*)(smem + BIG_OFF(q));   ld = 64; }
            else        { int u = t - 48, q = u >> 2, v2 = u & 3; i = v2 >> 1; j = v2 & 1;
                          Sb = (float*)(smem + SMALL_OFF(q)); ld = 32; }
            int g = lane >> 4, n = lane & 15;
            #pragma unroll
            for (int v = 0; v < 8; ++v) {
                int m = v + 8 * g;                      // C layout: VGPR v -> M=v / v+8
                Sb[(16 * i + m) * ld + (16 * j + n)] = acc[ti][v];
            }
        }
    }
    __syncthreads();

    // ---- masked row/col max-mean reductions for the six matrices ----
    for (int mi = 0; mi < 6; ++mi) {
        const float* S; int dim, n, m;
        switch (mi) {
            case 0: S = (const float*)(smem + BIG_OFF(0));   dim = 64; n = cnt_obj_p;  m = lsp;        break;
            case 1: S = (const float*)(smem + BIG_OFF(1));   dim = 64; n = cnt_obj_p;  m = lsn;        break;
            case 2: S = (const float*)(smem + BIG_OFF(2));   dim = 64; n = cnt_obj_n;  m = lsp;        break;
            case 3: S = (const float*)(smem + SMALL_OFF(0)); dim = 32; n = cnt_pred_p; m = cnt_rels_p; break;
            case 4: S = (const float*)(smem + SMALL_OFF(1)); dim = 32; n = cnt_pred_p; m = cnt_rels_n; break;
            default:S = (const float*)(smem + SMALL_OFF(2)); dim = 32; n = cnt_pred_n; m = cnt_rels_p; break;
        }
        if (tid < dim) {                               // row-max over valid columns
            float mx = NEGF;
            for (int c = 0; c < m; ++c) mx = fmaxf(mx, S[tid * dim + c]);
            s_scratch[tid] = (tid < n) ? mx : 0.f;
        } else if (tid >= 64 && tid < 64 + dim) {      // col-max over valid rows
            int c = tid - 64;
            float mx = NEGF;
            for (int r = 0; r < n; ++r) mx = fmaxf(mx, S[r * dim + c]);
            s_scratch[64 + c] = (c < m) ? mx : 0.f;
        }
        __syncthreads();
        if (tid == 0) {
            float sa = 0.f, sb = 0.f;
            for (int r = 0; r < dim; ++r) sa += s_scratch[r];
            for (int c = 0; c < dim; ++c) sb += s_scratch[64 + c];
            float va = (n > 0 && m > 0) ? 1.f : 0.f;
            s_res[2 * mi]     = va * sa / (float)(n > 0 ? n : 1);
            s_res[2 * mi + 1] = va * sb / (float)(m > 0 ? m : 1);
        }
        __syncthreads();
    }

    // ---- combine + hinge + atomic accumulate of mean ----
    if (tid == 0) {
        float S_ip_cp = s_res[0] + s_res[6];
        float S_cp_ip = s_res[1] + s_res[7];
        float S_ip_cn = s_res[2] + s_res[8];
        float S_cn_ip = s_res[3] + s_res[9];
        float S_in_cp = s_res[4] + s_res[10];
        float S_cp_in = s_res[5] + s_res[11];
        float cost = fmaxf(0.f, 1.f + S_ip_cn - S_ip_cp)
                   + fmaxf(0.f, 1.f + S_in_cp - S_ip_cp)
                   + fmaxf(0.f, 1.f + S_cn_ip - S_cp_ip)
                   + fmaxf(0.f, 1.f + S_cp_in - S_cp_ip);
        atomicAdd(out, cost * (1.0f / 256.0f));
    }
}

extern "C" void kernel_launch(void* const* d_in, const int* in_sizes, int n_in,
                              void* d_out, int out_size, void* d_ws, size_t ws_size,
                              hipStream_t stream) {
    (void)in_sizes; (void)n_in; (void)d_ws; (void)ws_size; (void)out_size;
    hipMemsetAsync(d_out, 0, sizeof(float), stream);
    matrix_loss_kernel<<<256, 256, 0, stream>>>(
        (const float*)d_in[0],  (const float*)d_in[1],
        (const int*)  d_in[2],  (const int*)  d_in[3],
        (const float*)d_in[4],  (const float*)d_in[5],
        (const int*)  d_in[6],  (const int*)  d_in[7],
        (const float*)d_in[8],  (const float*)d_in[9],
        (const int*)  d_in[10], (const int*)  d_in[11],
        (const float*)d_in[12], (const float*)d_in[13],
        (const int*)  d_in[14], (const int*)  d_in[15],
        (float*)d_out);
}
